// MiniGPT_730144440335
// MI455X (gfx1250) — matrix operand order repaired and checked by reading, not compiled, not run on hardware
//
#include <hip/hip_runtime.h>
#include <cstdint>
#include <cstddef>

typedef __attribute__((ext_vector_type(16))) __bf16 v16bf;
typedef __attribute__((ext_vector_type(8)))  float  v8f;
typedef __attribute__((ext_vector_type(4)))  unsigned int u32x4;
typedef __attribute__((ext_vector_type(8)))  int i32x8;
typedef __attribute__((ext_vector_type(4)))  int i32x4;

#if __has_builtin(__builtin_amdgcn_tensor_load_to_lds) && __has_builtin(__builtin_amdgcn_s_wait_tensorcnt)
#define USE_TDM 1
#else
#define USE_TDM 0
#endif

namespace cfg {
constexpr int B = 2, L = 1024, E = 1024, H = 16, V = 32000;
constexpr int FF = 4096, NEXP = 8, TOPK = 2, CAP = 512, NL = 2;
constexpr int HD = E / H;   // 64
constexpr int T = B * L;    // 2048
}

// ============ CDNA5 WMMA 16x16x32 bf16 fragment helpers ============
// A fragment (16x32) from row-major [M,K]:
//   lane r=lane&15 holds row r; half=lane>>4: elem i -> K=half*16+i
__device__ __forceinline__ v16bf frag_a_bf16(const __bf16* A, int lda, int row0, int k0, int lane) {
  const int half = (lane >> 4) & 1, r = lane & 15;
  const __bf16* p = A + (size_t)(row0 + r) * lda + k0;
  v16bf f;
#pragma unroll
  for (int i = 0; i < 16; ++i) {
    f[i] = p[half * 16 + i];
  }
  return f;
}

// B fragment (32x16): lane holds column n=lane&15, elem i -> K=k0+half*16+i.
// Source: row-major [N,K] bf16 (B[k][n] = W[n][k]).
__device__ __forceinline__ v16bf frag_b_bf16_nk(const __bf16* W, int ldw, int n0, int k0, int lane) {
  const int half = (lane >> 4) & 1, r = lane & 15;
  const __bf16* p = W + (size_t)(n0 + r) * ldw + k0 + half * 16;
  v16bf f;
#pragma unroll
  for (int i = 0; i < 16; ++i) f[i] = p[i];
  return f;
}

// B fragment from row-major bf16 [K,N]  (V-matrix in P*V)
__device__ __forceinline__ v16bf frag_b_bf16_kn(const __bf16* Bm, int ldb, int k0, int n0, int lane) {
  const int half = (lane >> 4) & 1, r = lane & 15;
  v16bf f;
#pragma unroll
  for (int i = 0; i < 16; ++i)
    f[i] = Bm[(size_t)(k0 + half * 16 + i) * ldb + n0 + r];
  return f;
}

__device__ __forceinline__ v8f wmma_bf16(v16bf a, v16bf b, v8f c) {
  return __builtin_amdgcn_wmma_f32_16x16x32_bf16(false, a, false, b, (short)0, c, false, false);
}

// ============ LDS staging helpers (blockDim.x == 128) ============
// Stage a 64x32 fp32 weight tile into LDS as bf16 (convert once per block).
__device__ __forceinline__ void stage_w_f32(const float* W, int ldw, int n0, int k0,
                                            __bf16* sW) {
  const int tid = threadIdx.x;
  const int row = tid >> 1, c0 = (tid & 1) * 16;
  const float* src = W + (size_t)(n0 + row) * ldw + k0 + c0;
  __bf16* dst = sW + row * 32 + c0;
#pragma unroll
  for (int i = 0; i < 16; ++i) dst[i] = (__bf16)src[i];
}

// Issue TDM load of a 64x32 bf16 activation tile into LDS (wave 0 only),
// overlapped with the weight staging; fall back to cooperative loads without TDM.
__device__ __forceinline__ void tile_a_issue(const __bf16* A, int lda, int m0, int k0,
                                             __bf16* sA) {
#if USE_TDM
  if (threadIdx.x < 32) {
    const unsigned long long ga = (unsigned long long)(uintptr_t)(A + (size_t)m0 * lda + k0);
    const unsigned ldsOff = (unsigned)(uintptr_t)sA;  // LDS aperture: low 32 bits = LDS offset
    u32x4 g0;
    g0[0] = 1u;                                                  // count = 1 (valid D#)
    g0[1] = ldsOff;                                              // lds_addr
    g0[2] = (unsigned)ga;                                        // global_addr[31:0]
    g0[3] = (unsigned)((ga >> 32) & 0x01FFFFFFull) | (2u << 30); // addr[56:32] | type=2
    i32x8 g1;
    g1[0] = (int)(1u << 16);                                     // data_size = 1 (2 bytes)
    g1[1] = (int)(((unsigned)lda & 0xFFFFu) << 16);              // tensor_dim0 lo16
    g1[2] = (int)((((unsigned)lda >> 16) & 0xFFFFu) |
                  (0xFFFFu << 16));                              // dim0 hi | tensor_dim1 lo16
    g1[3] = (int)(32u << 16);                                    // dim1 hi=0 | tile_dim0 = 32
    g1[4] = 64;                                                  // tile_dim1 = 64 (tile_dim2=0 -> 2D)
    g1[5] = lda;                                                 // tensor_dim0_stride[31:0]
    g1[6] = 0;
    g1[7] = 0;
    const i32x4 gz = {0, 0, 0, 0};
#if defined(__clang_major__) && __clang_major__ >= 23
    const i32x8 gz8 = {0, 0, 0, 0, 0, 0, 0, 0};
    __builtin_amdgcn_tensor_load_to_lds(g0, g1, gz, gz, gz8, 0);
#else
    __builtin_amdgcn_tensor_load_to_lds(g0, g1, gz, gz, 0);
#endif
  }
#else
  const int tid = threadIdx.x;
  const int row = tid >> 1, c0 = (tid & 1) * 16;
  const __bf16* src = A + (size_t)(m0 + row) * lda + k0 + c0;
  __bf16* dst = sA + row * 32 + c0;
#pragma unroll
  for (int i = 0; i < 16; ++i) dst[i] = src[i];
#endif
}

__device__ __forceinline__ void tile_a_wait() {
#if USE_TDM
  if (threadIdx.x < 32) __builtin_amdgcn_s_wait_tensorcnt(0);
#endif
}

// ============ Generic GEMM: Out[M,N](f32) = A[M,K](bf16) * W[N,K]^T(f32) (+Add) ============
// 128 threads (4 waves) per block; block computes a 64x64 tile.
// A tile arrives via TDM into LDS; W tile staged once per block as bf16 in LDS.
__global__ void gemm_kernel(const __bf16* __restrict__ A, const float* __restrict__ W,
                            float* __restrict__ Out, const float* __restrict__ Add,
                            int M, int N, int Kd,
                            long long aZ, long long wZ, long long oZ) {
  __shared__ __bf16 sA[64 * 32];
  __shared__ __bf16 sW[64 * 32];
  const int lane = threadIdx.x & 31;
  const int wave = threadIdx.x >> 5;
  const int z = blockIdx.z;
  A += (size_t)z * aZ; W += (size_t)z * wZ; Out += (size_t)z * oZ;
  if (Add) Add += (size_t)z * oZ;
  const int n0 = blockIdx.x * 64;
  const int m0 = blockIdx.y * 64;
  v8f acc[4] = {};
  for (int k0 = 0; k0 < Kd; k0 += 32) {
    tile_a_issue(A, Kd, m0, k0, sA);          // async DMA (TENSORcnt) or coop loads
    stage_w_f32(W, Kd, n0, k0, sW);           // fp32 -> bf16 convert while DMA runs
    __builtin_prefetch(W + (size_t)n0 * Kd + k0 + 512, 0, 0);  // global_prefetch_b8
    tile_a_wait();                            // s_wait_tensorcnt 0 (wave 0)
    __syncthreads();
    const v16bf a = frag_a_bf16(sA, 32, wave * 16, 0, lane);
#pragma unroll
    for (int t = 0; t < 4; ++t)
      acc[t] = wmma_bf16(a, frag_b_bf16_nk(sW, 32, t * 16, 0, lane), acc[t]);
    __syncthreads();
  }
  const int half = (lane >> 4) & 1, r = lane & 15;
  const int mw = m0 + wave * 16;
#pragma unroll
  for (int t = 0; t < 4; ++t)
#pragma unroll
    for (int j = 0; j < 8; ++j) {
      const size_t idx = (size_t)(mw + j + 8 * half) * N + n0 + t * 16 + r;
      float v = acc[t][j];
      if (Add) v += Add[idx];
      Out[idx] = v;
    }
  (void)M;
}

// ============ Fused expert FFN: H = silu(X*G^T) * (X*U^T), bf16 out ============
__global__ void ffn_gemm_kernel(const __bf16* __restrict__ X, const float* __restrict__ G,
                                const float* __restrict__ U, __bf16* __restrict__ Hout,
                                int Mc, int Nf, int Kd) {
  __shared__ __bf16 sA[64 * 32];
  __shared__ __bf16 sG[64 * 32];
  __shared__ __bf16 sU[64 * 32];
  const int lane = threadIdx.x & 31;
  const int wave = threadIdx.x >> 5;
  const int z = blockIdx.z;
  X += (size_t)z * Mc * Kd;
  G += (size_t)z * Nf * Kd;
  U += (size_t)z * Nf * Kd;
  Hout += (size_t)z * Mc * Nf;
  const int n0 = blockIdx.x * 64;
  const int m0 = blockIdx.y * 64;
  v8f ag[4] = {}, au[4] = {};
  for (int k0 = 0; k0 < Kd; k0 += 32) {
    tile_a_issue(X, Kd, m0, k0, sA);
    stage_w_f32(G, Kd, n0, k0, sG);
    stage_w_f32(U, Kd, n0, k0, sU);
    tile_a_wait();
    __syncthreads();
    const v16bf a = frag_a_bf16(sA, 32, wave * 16, 0, lane);
#pragma unroll
    for (int t = 0; t < 4; ++t) {
      ag[t] = wmma_bf16(a, frag_b_bf16_nk(sG, 32, t * 16, 0, lane), ag[t]);
      au[t] = wmma_bf16(a, frag_b_bf16_nk(sU, 32, t * 16, 0, lane), au[t]);
    }
    __syncthreads();
  }
  const int half = (lane >> 4) & 1, r = lane & 15;
  const int mw = m0 + wave * 16;
#pragma unroll
  for (int t = 0; t < 4; ++t)
#pragma unroll
    for (int j = 0; j < 8; ++j) {
      const float g = ag[t][j];
      const float h = (g / (1.f + expf(-g))) * au[t][j];
      Hout[(size_t)(mw + j + 8 * half) * Nf + n0 + t * 16 + r] = (__bf16)h;
    }
}

// ============ Fused causal flash attention, one wave per (b, h, 16-row q tile) ============
__global__ void attn_kernel(const __bf16* __restrict__ Q, const __bf16* __restrict__ Kb,
                            const __bf16* __restrict__ Vb, __bf16* __restrict__ Y) {
  using namespace cfg;
  __shared__ __bf16 pls[16 * 32];
  const int lane = threadIdx.x & 31;
  const int half = (lane >> 4) & 1, r = lane & 15;
  const int q0 = blockIdx.x * 16;
  const int h  = blockIdx.y;
  const int b  = blockIdx.z;
  const size_t base = ((size_t)b * L) * E + h * HD;
  const __bf16* Qp = Q + base;
  const __bf16* Kp = Kb + base;
  const __bf16* Vp = Vb + base;
  const v16bf qf0 = frag_a_bf16(Qp, E, q0, 0, lane);
  const v16bf qf1 = frag_a_bf16(Qp, E, q0, 32, lane);
  float m[8], s[8];
  v8f o[4] = {};
#pragma unroll
  for (int j = 0; j < 8; ++j) { m[j] = -3.0e38f; s[j] = 0.f; }
  for (int k0 = 0; k0 <= q0 + 15; k0 += 32) {
    v8f sc0 = {}, sc1 = {};
    sc0 = wmma_bf16(qf0, frag_b_bf16_nk(Kp, E, k0, 0, lane), sc0);
    sc0 = wmma_bf16(qf1, frag_b_bf16_nk(Kp, E, k0, 32, lane), sc0);
    sc1 = wmma_bf16(qf0, frag_b_bf16_nk(Kp, E, k0 + 16, 0, lane), sc1);
    sc1 = wmma_bf16(qf1, frag_b_bf16_nk(Kp, E, k0 + 16, 32, lane), sc1);
#pragma unroll
    for (int j = 0; j < 8; ++j) {
      const int qi = q0 + j + 8 * half;
      sc0[j] = (k0 + r      <= qi) ? sc0[j] * 0.125f : -3.0e38f;  // 1/sqrt(64)
      sc1[j] = (k0 + 16 + r <= qi) ? sc1[j] * 0.125f : -3.0e38f;
    }
#pragma unroll
    for (int j = 0; j < 8; ++j) {
      float v = fmaxf(sc0[j], sc1[j]);
#pragma unroll
      for (int off = 1; off < 16; off <<= 1) v = fmaxf(v, __shfl_xor(v, off, 32));
      const float nm = fmaxf(m[j], v);
      const float p0 = expf(sc0[j] - nm);
      const float p1 = expf(sc1[j] - nm);
      float rs = p0 + p1;
#pragma unroll
      for (int off = 1; off < 16; off <<= 1) rs += __shfl_xor(rs, off, 32);
      const float scale = expf(m[j] - nm);
      s[j] = s[j] * scale + rs;
      m[j] = nm;
#pragma unroll
      for (int nt = 0; nt < 4; ++nt) o[nt][j] *= scale;
      pls[(j + 8 * half) * 32 + r]      = (__bf16)p0;  // re-stage P: C-layout -> A-layout
      pls[(j + 8 * half) * 32 + 16 + r] = (__bf16)p1;
    }
    __syncthreads();
    const v16bf pa = frag_a_bf16(pls, 32, 0, 0, lane);
#pragma unroll
    for (int nt = 0; nt < 4; ++nt)
      o[nt] = wmma_bf16(pa, frag_b_bf16_kn(Vp, E, k0, nt * 16, lane), o[nt]);
    __syncthreads();
  }
#pragma unroll
  for (int nt = 0; nt < 4; ++nt)
#pragma unroll
    for (int j = 0; j < 8; ++j)
      Y[base + (size_t)(q0 + j + 8 * half) * E + nt * 16 + r] = (__bf16)(o[nt][j] / s[j]);
}

// ============ Elementwise / small kernels ============
__global__ void embed_kernel(const int* __restrict__ tok, const float* __restrict__ emb,
                             float* __restrict__ X) {
  using namespace cfg;
  const int t = blockIdx.x;
  const size_t src = (size_t)tok[t] * E, dst = (size_t)t * E;
  for (int i = threadIdx.x; i < E; i += blockDim.x) X[dst + i] = emb[src + i];
}

__global__ void rmsnorm_kernel(const float* __restrict__ X, const float* __restrict__ w,
                               __bf16* __restrict__ Out) {
  using namespace cfg;
  __shared__ float red[256];
  const int t = blockIdx.x, tid = threadIdx.x;
  const float* xr = X + (size_t)t * E;
  float ss = 0.f;
  for (int i = tid; i < E; i += 256) { const float v = xr[i]; ss += v * v; }
  red[tid] = ss;
  __syncthreads();
  for (int st = 128; st > 0; st >>= 1) {
    if (tid < st) red[tid] += red[tid + st];
    __syncthreads();
  }
  const float inv = rsqrtf(red[0] / (float)E + 1e-6f);
  for (int i = tid; i < E; i += 256) Out[(size_t)t * E + i] = (__bf16)(xr[i] * inv * w[i]);
}

__global__ void rope_cvt_kernel(const float* __restrict__ qf, const float* __restrict__ kf,
                                const float* __restrict__ vf, __bf16* __restrict__ qb,
                                __bf16* __restrict__ kb, __bf16* __restrict__ vb) {
  using namespace cfg;
  const int t = blockIdx.x;
  const int l = t % L;
  const size_t base = (size_t)t * E;
  for (int p = threadIdx.x; p < E / 2; p += blockDim.x) {
    const int j = p & (HD / 2 - 1);
    const float fr = (float)l * powf(10000.0f, -(float)(2 * j) / (float)HD);
    const float c = cosf(fr), sn = sinf(fr);
    const size_t i0 = base + 2 * p, i1 = i0 + 1;
    const float qr = qf[i0], qi = qf[i1];
    qb[i0] = (__bf16)(qr * c - qi * sn);
    qb[i1] = (__bf16)(qr * sn + qi * c);
    const float kr = kf[i0], ki = kf[i1];
    kb[i0] = (__bf16)(kr * c - ki * sn);
    kb[i1] = (__bf16)(kr * sn + ki * c);
    vb[i0] = (__bf16)vf[i0];
    vb[i1] = (__bf16)vf[i1];
  }
}

__global__ void router_kernel(const __bf16* __restrict__ Xn, const float* __restrict__ Wr,
                              const float* __restrict__ br, float* __restrict__ scores,
                              int* __restrict__ tki, float* __restrict__ tkv) {
  using namespace cfg;
  __shared__ float red[256 * NEXP];
  const int t = blockIdx.x, tid = threadIdx.x;
  const __bf16* xr = Xn + (size_t)t * E;
  float acc[NEXP];
#pragma unroll
  for (int n = 0; n < NEXP; ++n) acc[n] = 0.f;
  for (int i = tid; i < E; i += 256) {
    const float xv = (float)xr[i];
#pragma unroll
    for (int n = 0; n < NEXP; ++n) acc[n] += xv * Wr[(size_t)n * E + i];
  }
#pragma unroll
  for (int n = 0; n < NEXP; ++n) red[tid * NEXP + n] = acc[n];
  __syncthreads();
  for (int st = 128; st > 0; st >>= 1) {
    if (tid < st)
#pragma unroll
      for (int n = 0; n < NEXP; ++n) red[tid * NEXP + n] += red[(tid + st) * NEXP + n];
    __syncthreads();
  }
  if (tid == 0) {
    float sc[NEXP];
    float mx = -3.0e38f;
    for (int n = 0; n < NEXP; ++n) { sc[n] = red[n] + br[n]; mx = fmaxf(mx, sc[n]); }
    float ssum = 0.f;
    for (int n = 0; n < NEXP; ++n) { sc[n] = expf(sc[n] - mx); ssum += sc[n]; }
    for (int n = 0; n < NEXP; ++n) { sc[n] /= ssum; scores[(size_t)t * NEXP + n] = sc[n]; }
    int i0 = 0;
    for (int n = 1; n < NEXP; ++n) if (sc[n] > sc[i0]) i0 = n;   // low index wins ties
    int i1 = (i0 == 0) ? 1 : 0;
    for (int n = 0; n < NEXP; ++n) if (n != i0 && sc[n] > sc[i1]) i1 = n;
    const float tot = sc[i0] + sc[i1];
    tki[t * 2] = i0; tki[t * 2 + 1] = i1;
    tkv[t * 2] = sc[i0] / tot; tkv[t * 2 + 1] = sc[i1] / tot;
  }
}

// Rank within expert = # earlier slots with same expert (stable argsort semantics).
__global__ void rank_kernel(const int* __restrict__ tki, int* __restrict__ rank,
                            int* __restrict__ counts) {
  using namespace cfg;
  if (threadIdx.x == 0 && blockIdx.x == 0) {
    int cnt[NEXP];
    for (int n = 0; n < NEXP; ++n) cnt[n] = 0;
    for (int rr = 0; rr < T * TOPK; ++rr) { const int e = tki[rr]; rank[rr] = cnt[e]++; }
    for (int n = 0; n < NEXP; ++n) counts[n] = cnt[n];
  }
}

__global__ void zero_bf16_kernel(__bf16* p, size_t n) {
  size_t i = (size_t)blockIdx.x * blockDim.x + threadIdx.x;
  const size_t st = (size_t)gridDim.x * blockDim.x;
  for (; i < n; i += st) p[i] = (__bf16)0.f;
}
__global__ void zero_f32_kernel(float* p, size_t n) {
  size_t i = (size_t)blockIdx.x * blockDim.x + threadIdx.x;
  const size_t st = (size_t)gridDim.x * blockDim.x;
  for (; i < n; i += st) p[i] = 0.f;
}
__global__ void cvt_f32_bf16_kernel(const float* __restrict__ in, __bf16* __restrict__ out, size_t n) {
  size_t i = (size_t)blockIdx.x * blockDim.x + threadIdx.x;
  const size_t st = (size_t)gridDim.x * blockDim.x;
  for (; i < n; i += st) out[i] = (__bf16)in[i];
}

__global__ void dispatch_kernel(const __bf16* __restrict__ Xn, const int* __restrict__ tki,
                                const int* __restrict__ rank, __bf16* __restrict__ Xnce) {
  using namespace cfg;
  const int rr = blockIdx.x;
  const int e = tki[rr], rk = rank[rr];
  if (rk >= CAP) return;
  const __bf16* src = Xn + (size_t)(rr >> 1) * E;   // token = slot / TOPK
  __bf16* dst = Xnce + ((size_t)e * CAP + rk) * E;
  for (int i = threadIdx.x; i < E; i += blockDim.x) dst[i] = src[i];
}

__global__ void combine_kernel(float* __restrict__ X, const float* __restrict__ out2,
                               const int* __restrict__ tki, const int* __restrict__ rank,
                               const float* __restrict__ tkv) {
  using namespace cfg;
  const int t = blockIdx.x;
  float* xr = X + (size_t)t * E;
  for (int i = threadIdx.x; i < E; i += blockDim.x) {
    float add = 0.f;
#pragma unroll
    for (int k = 0; k < TOPK; ++k) {
      const int rr = t * TOPK + k;
      const int rk = rank[rr];
      if (rk < CAP) add += tkv[rr] * out2[((size_t)tki[rr] * CAP + rk) * E + i];
    }
    xr[i] += add;
  }
}

__global__ void loss_kernel(const float* __restrict__ scores, const int* __restrict__ counts,
                            float* __restrict__ loss) {
  using namespace cfg;
  __shared__ float red[256 * NEXP];
  const int tid = threadIdx.x;
  float acc[NEXP];
#pragma unroll
  for (int n = 0; n < NEXP; ++n) acc[n] = 0.f;
  for (int t = tid; t < T; t += 256)
#pragma unroll
    for (int n = 0; n < NEXP; ++n) acc[n] += scores[(size_t)t * NEXP + n];
#pragma unroll
  for (int n = 0; n < NEXP; ++n) red[tid * NEXP + n] = acc[n];
  __syncthreads();
  for (int st = 128; st > 0; st >>= 1) {
    if (tid < st)
#pragma unroll
      for (int n = 0; n < NEXP; ++n) red[tid * NEXP + n] += red[(tid + st) * NEXP + n];
    __syncthreads();
  }
  if (tid == 0) {
    float aux = 0.f;
    for (int n = 0; n < NEXP; ++n) {
      const float imp = red[n] / (float)T;
      const float load = (float)counts[n] / (float)(T * TOPK);
      aux += load * imp;
    }
    loss[0] += (float)NEXP * aux;
  }
}

// ============ Host-side orchestration ============
extern "C" void kernel_launch(void* const* d_in, const int* in_sizes, int n_in,
                              void* d_out, int out_size, void* d_ws, size_t ws_size,
                              hipStream_t stream) {
  using namespace cfg;
  (void)in_sizes; (void)n_in; (void)out_size; (void)ws_size;

  const int*   tokens = (const int*)d_in[0];
  const float* emb    = (const float*)d_in[1];
  const float* out_w  = (const float*)d_in[2];

  char* ws = (char*)d_ws;
  auto carve = [&](size_t bytes) -> char* {
    char* r = ws;
    ws += (bytes + 255) & ~(size_t)255;
    return r;
  };
  float*  xf     = (float*) carve(sizeof(float)  * (size_t)T * E);
  __bf16* xn     = (__bf16*)carve(sizeof(__bf16) * (size_t)T * E);
  float*  qf     = (float*) carve(sizeof(float)  * (size_t)T * E);
  float*  kf     = (float*) carve(sizeof(float)  * (size_t)T * E);
  float*  vf     = (float*) carve(sizeof(float)  * (size_t)T * E);
  __bf16* qb     = (__bf16*)carve(sizeof(__bf16) * (size_t)T * E);
  __bf16* kb     = (__bf16*)carve(sizeof(__bf16) * (size_t)T * E);
  __bf16* vb     = (__bf16*)carve(sizeof(__bf16) * (size_t)T * E);
  __bf16* yb     = (__bf16*)carve(sizeof(__bf16) * (size_t)T * E);
  float*  scores = (float*) carve(sizeof(float)  * (size_t)T * NEXP);
  int*    tki    = (int*)   carve(sizeof(int)    * (size_t)T * TOPK);
  float*  tkv    = (float*) carve(sizeof(float)  * (size_t)T * TOPK);
  int*    rankb  = (int*)   carve(sizeof(int)    * (size_t)T * TOPK);
  int*    counts = (int*)   carve(sizeof(int)    * 64);
  __bf16* xnce   = (__bf16*)carve(sizeof(__bf16) * (size_t)NEXP * CAP * E);
  __bf16* hbuf   = (__bf16*)carve(sizeof(__bf16) * (size_t)NEXP * CAP * FF);
  float*  out2   = (float*) carve(sizeof(float)  * (size_t)NEXP * CAP * E);

  float* lossp = (float*)d_out + (size_t)T * V;
  zero_f32_kernel<<<1, 32, 0, stream>>>(lossp, 1);
  embed_kernel<<<T, 256, 0, stream>>>(tokens, emb, xf);

  const dim3 gProj(E / 64, T / 64, 1);
  for (int li = 0; li < NL; ++li) {
    const int base = 3 + li * 11;
    const float* qw   = (const float*)d_in[base + 0];
    const float* kw   = (const float*)d_in[base + 1];
    const float* vw   = (const float*)d_in[base + 2];
    const float* ow   = (const float*)d_in[base + 3];
    const float* rms1 = (const float*)d_in[base + 4];
    const float* rms2 = (const float*)d_in[base + 5];
    const float* rw   = (const float*)d_in[base + 6];
    const float* rb   = (const float*)d_in[base + 7];
    const float* gate = (const float*)d_in[base + 8];
    const float* enc  = (const float*)d_in[base + 9];
    const float* dec  = (const float*)d_in[base + 10];

    rmsnorm_kernel<<<T, 256, 0, stream>>>(xf, rms1, xn);
    gemm_kernel<<<gProj, 128, 0, stream>>>(xn, qw, qf, nullptr, T, E, E, 0, 0, 0);
    gemm_kernel<<<gProj, 128, 0, stream>>>(xn, kw, kf, nullptr, T, E, E, 0, 0, 0);
    gemm_kernel<<<gProj, 128, 0, stream>>>(xn, vw, vf, nullptr, T, E, E, 0, 0, 0);
    rope_cvt_kernel<<<T, 256, 0, stream>>>(qf, kf, vf, qb, kb, vb);
    attn_kernel<<<dim3(L / 16, H, B), 32, 0, stream>>>(qb, kb, vb, yb);
    gemm_kernel<<<gProj, 128, 0, stream>>>(yb, ow, xf, xf, T, E, E, 0, 0, 0);  // x += y @ ow^T

    rmsnorm_kernel<<<T, 256, 0, stream>>>(xf, rms2, xn);
    router_kernel<<<T, 256, 0, stream>>>(xn, rw, rb, scores, tki, tkv);
    rank_kernel<<<1, 32, 0, stream>>>(tki, rankb, counts);
    zero_bf16_kernel<<<2048, 256, 0, stream>>>(xnce, (size_t)NEXP * CAP * E);
    dispatch_kernel<<<T * TOPK, 256, 0, stream>>>(xn, tki, rankb, xnce);
    ffn_gemm_kernel<<<dim3(FF / 64, CAP / 64, NEXP), 128, 0, stream>>>(xnce, gate, enc, hbuf,
                                                                       CAP, FF, E);
    gemm_kernel<<<dim3(E / 64, CAP / 64, NEXP), 128, 0, stream>>>(
        hbuf, dec, out2, nullptr, CAP, E, FF,
        (long long)CAP * FF, (long long)E * FF, (long long)CAP * E);
    combine_kernel<<<T, 256, 0, stream>>>(xf, out2, tki, rankb, tkv);
    loss_kernel<<<1, 256, 0, stream>>>(scores, counts, lossp);
  }

  cvt_f32_bf16_kernel<<<2048, 256, 0, stream>>>(xf, xn, (size_t)T * E);
  gemm_kernel<<<dim3(V / 64, T / 64, 1), 128, 0, stream>>>(xn, out_w, (float*)d_out, nullptr,
                                                           T, V, E, 0, 0, 0);
}
